// TransformerBlockQuantum_65481071405192
// MI455X (gfx1250) — matrix operand order repaired and checked by reading, not compiled, not run on hardware
//
#include <hip/hip_runtime.h>
#include <hip/hip_bf16.h>

typedef _Float16 v16h __attribute__((ext_vector_type(16)));
typedef _Float16 v8h  __attribute__((ext_vector_type(8)));
typedef float    v8f  __attribute__((ext_vector_type(8)));
typedef float    v4f  __attribute__((ext_vector_type(4)));

#define ECOLS   128      // embed dim E
#define FDIM    2048     // FFN dim F
#define ROWS    64       // rows per workgroup (4 M-tiles)
#define NWAVES  8        // 256 threads
#define CHUNK   256      // K-chunk of GEMM2 (= N-chunk of GEMM1)
#define CHUNKP  264      // padded LDS row stride (halves) -> conflict-free b128 reads
#define NCHUNK  8        // FDIM / CHUNK
#define LN_EPS  1e-5f

// ---- weight precision prep: f32 -> f16 copies in workspace (L2-resident) ----
__global__ void qtb_prep_weights(const float* __restrict__ w1, const float* __restrict__ w2,
                                 _Float16* __restrict__ w1h, _Float16* __restrict__ w2h) {
  int i = blockIdx.x * 256 + threadIdx.x;
  if (i < ECOLS * FDIM) w2h[i] = (_Float16)w2[i];
  if (i < FDIM * 8)     w1h[i] = (_Float16)w1[i];
}

// ---- fused quantum transformer block: attn+LN1+q -> WMMA GEMM1 -> WMMA GEMM2 -> LN2 ----
__launch_bounds__(256)
__global__ void qtb_kernel(const float* __restrict__ x,
                           const float* __restrict__ theta,
                           const float* __restrict__ phi,
                           const _Float16* __restrict__ w1h,
                           const float* __restrict__ b1,
                           const _Float16* __restrict__ w2h,
                           const float* __restrict__ b2,
                           const float* __restrict__ g1,
                           const float* __restrict__ be1,
                           const float* __restrict__ g2,
                           const float* __restrict__ be2,
                           float* __restrict__ out) {
  __shared__ float    x1s[ROWS * ECOLS];        // 32 KB: x1 rows, later reused as (x1+ffn+b2)
  __shared__ _Float16 hbuf[2][ROWS * CHUNKP];   // 66 KB: double-buffered h tile (f16)
  __shared__ _Float16 qbuf[ROWS * 8];           // 1 KB: q rows (f16)

  const int tid  = threadIdx.x;
  const int wave = tid >> 5;      // 0..7
  const int lane = tid & 31;
  const int l16  = lane & 15;
  const int hi   = lane >> 4;     // 0/1 half-wave
  const long rowBase = (long)blockIdx.x * ROWS;

  // ================= Phase 1: quantum attn + LN1 + quantum q =================
  {
    float th[8], ph[8];
#pragma unroll
    for (int k = 0; k < 8; ++k) { th[k] = theta[k]; ph[k] = phi[k]; }

    for (int it = 0; it < 4; ++it) {
      const int lrow = wave * 8 + it * 2 + hi;            // local row, lane is head l16
      const float* xr = x + (rowBase + lrow) * ECOLS + l16 * 8;
      float xv[8];
      v4f xa = *(const v4f*)xr;
      v4f xb = *(const v4f*)(xr + 4);
#pragma unroll
      for (int j = 0; j < 4; ++j) { xv[j] = xa[j]; xv[4 + j] = xb[j]; }

      // cumulative product of cosines within the head
      float cps[8], cp = 1.0f;
#pragma unroll
      for (int k = 0; k < 8; ++k) { cp *= __cosf(xv[k] + th[k]); cps[k] = cp; }
      float yv[8];
      yv[0] = xv[0] + cps[7];                              // wire0 gets full product
#pragma unroll
      for (int k = 1; k < 8; ++k) yv[k] = xv[k] + cps[k];

      // LN1: reduce across the 16 lanes of this row-half
      float s = 0.f, ss = 0.f;
#pragma unroll
      for (int k = 0; k < 8; ++k) { s += yv[k]; ss += yv[k] * yv[k]; }
      for (int d = 8; d > 0; d >>= 1) { s += __shfl_xor(s, d, 16); ss += __shfl_xor(ss, d, 16); }
      const float mean = s * (1.0f / ECOLS);
      const float var  = ss * (1.0f / ECOLS) - mean * mean;
      const float rs   = rsqrtf(var + LN_EPS);

      const int e0 = l16 * 8;
      float x1a[8];
#pragma unroll
      for (int k = 0; k < 8; ++k) x1a[k] = (yv[k] - mean) * rs * g1[e0 + k] + be1[e0 + k];
#pragma unroll
      for (int k = 0; k < 8; ++k) x1s[lrow * ECOLS + e0 + k] = x1a[k];

      if (l16 == 0) {                                      // head0 lane owns x1[:8] -> q
#pragma unroll
        for (int j = 0; j < 8; ++j)
          qbuf[lrow * 8 + j] = (_Float16)(__cosf(ph[j]) * __cosf(x1a[j]));
      }
    }
  }
  __syncthreads();

  // ================= Phase 2: WMMA GEMM1 (q@w1^T, K=8 padded to 32) + GEMM2 (h@w2^T) =========
  // GEMM1: wave w produces M-tile (w&3), f-tiles (w>>2)*8 .. +7 of each chunk.
  // GEMM2: wave w owns N-tile w (cols 16w..16w+15) for ALL 4 M-tiles -> each global
  //        B-fragment feeds 4 WMMAs; w2 is read exactly once per workgroup.
  const int mt1 = wave & 3;       // GEMM1 M-tile owned by this wave
  const int fg  = wave >> 2;      // GEMM1 f-tile group (0/1)

  // GEMM1 A-fragment (constant): A = q, 16x32 with K=8 real. Lane layout (16-bit A 16x32):
  // lanes 0-15 hold M=lane, K=0..7 in halves 0..7 (rest zero); lanes 16-31 hold K=8..15/24..31 -> zero.
  v16h aq;
#pragma unroll
  for (int j = 0; j < 16; ++j) aq[j] = (_Float16)0;
  if (lane < 16) {
    v8h qv = *(const v8h*)(qbuf + (mt1 * 16 + lane) * 8);
#pragma unroll
    for (int j = 0; j < 8; ++j) aq[j] = qv[j];
  }

  v8f acc[4];                     // acc[mtile] for N-tile = wave
#pragma unroll
  for (int t = 0; t < 4; ++t)
#pragma unroll
    for (int j = 0; j < 8; ++j) acc[t][j] = 0.0f;

  const int eCol = wave * 16 + l16;                 // this lane's output column (B col n=l16)

  for (int c = 0; c < NCHUNK; ++c) {
    _Float16* hb = hbuf[c & 1];

    // speculative prefetch of next chunk's w2 row segment (dropped if OOB translation)
    if (c + 1 < NCHUNK)
      __builtin_prefetch(w2h + (long)eCol * FDIM + (c + 1) * CHUNK, 0, 0);

    // ---- GEMM1: h[mt1*16..+15][8 f-tiles] of the chunk ----
#pragma unroll
    for (int t = 0; t < 8; ++t) {
      const int ft = fg * 8 + t;                        // f-tile within chunk (0..15)
      const int f  = c * CHUNK + ft * 16 + l16;         // this lane's output column f
      // B = w1^T (32x16, K=8 real): lanes 0-15 col n=lane, K=0..7 in halves 0..7; rest zero.
      v16h bw;
#pragma unroll
      for (int j = 0; j < 16; ++j) bw[j] = (_Float16)0;
      if (lane < 16) {
        v8h wv = *(const v8h*)(w1h + f * 8);
#pragma unroll
        for (int j = 0; j < 8; ++j) bw[j] = wv[j];
      }
      v8f cz;
#pragma unroll
      for (int j = 0; j < 8; ++j) cz[j] = 0.0f;
      v8f hacc = __builtin_amdgcn_wmma_f32_16x16x32_f16(false, aq, false, bw,
                                                        (short)0, cz, false, false);
      // bias + relu + f16 -> LDS h tile (row-major, padded stride)
      const float bb   = b1[f];
      const int   mrow = mt1 * 16 + hi * 8;             // D: vgpr r -> M = r + 8*hi
#pragma unroll
      for (int r = 0; r < 8; ++r) {
        float hv = hacc[r] + bb;
        hv = hv > 0.0f ? hv : 0.0f;
        hb[(mrow + r) * CHUNKP + ft * 16 + l16] = (_Float16)hv;
      }
    }
    __syncthreads();   // h chunk visible (double-buffer -> one barrier per chunk is safe)

    // ---- GEMM2: 8 K-steps; per step: 1 global B-frag, 4 LDS A-frags, 4 WMMAs ----
#pragma unroll 2
    for (int ks = 0; ks < 8; ++ks) {
      const int kk = c * CHUNK + ks * 32 + hi * 16;     // lanes 0-15: K lo 16, 16-31: K hi 16
      v16h bw2 = *(const v16h*)(w2h + (long)eCol * FDIM + kk);
      const int k0 = ks * 32 + hi * 16;                 // A-frag K offsets (+0..15, contiguous)
#pragma unroll
      for (int t = 0; t < 4; ++t) {
        const int m = t * 16 + l16;                     // A row for M-tile t
        v8h a0 = *(const v8h*)(hb + m * CHUNKP + k0);
        v8h a1 = *(const v8h*)(hb + m * CHUNKP + k0 + 8);
        v16h ah;
#pragma unroll
        for (int j = 0; j < 8; ++j) { ah[j] = a0[j]; ah[8 + j] = a1[j]; }
        acc[t] = __builtin_amdgcn_wmma_f32_16x16x32_f16(false, ah, false, bw2,
                                                        (short)0, acc[t], false, false);
      }
    }
  }

  // ================= Epilogue: +b2, +x1 residual (in place), LN2, store ======================
  {
    const float bb = b2[eCol];
#pragma unroll
    for (int t = 0; t < 4; ++t) {
      const int mrow = t * 16 + hi * 8;
#pragma unroll
      for (int r = 0; r < 8; ++r)
        x1s[(mrow + r) * ECOLS + eCol] += acc[t][r] + bb;   // y = x1 + ffn + b2
    }
  }
  __syncthreads();

  {
    v4f gg = *(const v4f*)(g2  + lane * 4);
    v4f bv = *(const v4f*)(be2 + lane * 4);
    for (int rr = 0; rr < 8; ++rr) {
      const int m = wave * 8 + rr;
      v4f yv = *(const v4f*)(x1s + m * ECOLS + lane * 4);
      float s  = yv[0] + yv[1] + yv[2] + yv[3];
      float ss = yv[0]*yv[0] + yv[1]*yv[1] + yv[2]*yv[2] + yv[3]*yv[3];
      for (int d = 16; d > 0; d >>= 1) { s += __shfl_xor(s, d, 32); ss += __shfl_xor(ss, d, 32); }
      const float mean = s * (1.0f / ECOLS);
      const float var  = ss * (1.0f / ECOLS) - mean * mean;
      const float rs   = rsqrtf(var + LN_EPS);
      v4f res;
#pragma unroll
      for (int j = 0; j < 4; ++j) res[j] = (yv[j] - mean) * rs * gg[j] + bv[j];
      *(v4f*)(out + (rowBase + m) * ECOLS + lane * 4) = res;
    }
  }
}

extern "C" void kernel_launch(void* const* d_in, const int* in_sizes, int n_in,
                              void* d_out, int out_size, void* d_ws, size_t ws_size,
                              hipStream_t stream) {
  const float* x     = (const float*)d_in[0];
  const float* theta = (const float*)d_in[1];
  const float* phi   = (const float*)d_in[2];
  const float* w1    = (const float*)d_in[3];
  const float* b1    = (const float*)d_in[4];
  const float* w2    = (const float*)d_in[5];
  const float* b2    = (const float*)d_in[6];
  const float* g1    = (const float*)d_in[7];
  const float* be1   = (const float*)d_in[8];
  const float* g2    = (const float*)d_in[9];
  const float* be2   = (const float*)d_in[10];
  float* out = (float*)d_out;

  _Float16* w2h = (_Float16*)d_ws;                                     // 128*2048 halves
  _Float16* w1h = (_Float16*)((char*)d_ws + (size_t)ECOLS * FDIM * 2); // 2048*8 halves

  // f32 -> f16 weight copies (one pass; L2-resident afterwards)
  qtb_prep_weights<<<(ECOLS * FDIM + 255) / 256, 256, 0, stream>>>(w1, w2, w1h, w2h);

  const int totalRows = 8 * 4096;               // B * S
  qtb_kernel<<<totalRows / ROWS, 256, 0, stream>>>(x, theta, phi, w1h, b1, w2h, b2,
                                                   g1, be1, g2, be2, out);
}
